// MLP_74637941669966
// MI455X (gfx1250) — matrix operand order repaired and checked by reading, not compiled, not run on hardware
//
#include <hip/hip_runtime.h>
#include <stdint.h>

// ---------------- CDNA5 types ----------------
typedef __bf16 v16bf __attribute__((ext_vector_type(16)));
typedef float  v8f   __attribute__((ext_vector_type(8)));
typedef unsigned int v4u __attribute__((ext_vector_type(4)));
typedef int v8i __attribute__((ext_vector_type(8)));
typedef int v4i __attribute__((ext_vector_type(4)));

union Frag {
    uint32_t u[8];
    v16bf    v;
};

#define THREADS 256      // 8 wave32
#define PTS     32       // points per workgroup
#define ROWS    128      // PTS * 4 shifts = GEMM rows
#define XSTR    264      // bf16 stride of activation rows: 528B = 16B aligned, 4-bank spread/row
#define KSTR    264      // bf16 stride of transposed weight rows (matches TDM pad: 512B + 16B)
#define HDIM    256
#define WDIM    256

// LDS layout (bytes)
#define X0_OFF     0
#define X1_OFF     (ROWS * XSTR * 2)                 // 67584
#define LW_OFF     (2 * ROWS * XSTR * 2)             // 135168
#define AREA_OFF   (LW_OFF + 256 * KSTR * 2)         // 270336
#define PRED_OFF   (AREA_OFF + ROWS * 4)             // 270848
#define SMEM_BYTES (PRED_OFF + ROWS * 3 * 4)         // 272384  (< 320KB WGP LDS)

#define WS_LAYER_ELEMS 65536                         // 256 n x 256 k bf16 per layer

// Native bf16 converts (gfx1250 has hardware cvt; clang emits it for __bf16 casts)
__device__ __forceinline__ uint16_t f2bf(float f) {
    __bf16 b = (__bf16)f;
    return __builtin_bit_cast(uint16_t, b);
}
__device__ __forceinline__ float bf2f(uint16_t h) {
    return (float)__builtin_bit_cast(__bf16, h);
}

// ---------------- Pre-pass: fp32 W[k][256] -> bf16 ws[layer][n][256k] (transposed, zero-padded K)
__global__ void __launch_bounds__(THREADS)
prep_weights_kernel(const float* __restrict__ w0, const float* __restrict__ w1,
                    const float* __restrict__ w2, const float* __restrict__ w3,
                    uint16_t* __restrict__ ws) {
    int idx = blockIdx.x * THREADS + threadIdx.x;    // 0 .. 4*65536-1
    int l   = idx >> 16;
    int rem = idx & 65535;
    int k   = rem >> 8;                              // coalesced global reads over n
    int n   = rem & 255;
    const float* W = (l == 0) ? w0 : (l == 1) ? w1 : (l == 2) ? w2 : w3;
    int Kreal = (l == 0) ? 68 : 256;
    float v = (k < Kreal) ? W[k * 256 + n] : 0.0f;
    ws[l * WS_LAYER_ELEMS + n * 256 + k] = f2bf(v);
}

// ---------------- TDM: DMA one 256 x 512B weight tile into LDS with 16B/row padding ----------
__device__ __forceinline__ void stage_weights_tdm(const uint16_t* __restrict__ gsrc,
                                                  uint16_t* __restrict__ lw, int tid) {
#if __has_builtin(__builtin_amdgcn_tensor_load_to_lds) && __has_builtin(__builtin_amdgcn_s_wait_tensorcnt)
    (void)lw;
    if ((tid >> 5) == 0) {                           // wave 0 issues the DMA (EXEC-independent)
        uint64_t ga = (uint64_t)(uintptr_t)gsrc;
        v4u g0;
        g0[0] = 1u;                                  // count=1, user descriptor
        g0[1] = (unsigned)LW_OFF;                    // lds_addr (bytes)
        g0[2] = (unsigned)(ga & 0xFFFFFFFFu);        // global_addr[31:0]
        g0[3] = (unsigned)((ga >> 32) & 0x01FFFFFFu) | (2u << 30);  // addr[56:32] | type=2
        v8i g1;
        g1[0] = (1 << 16)                            // data_size = 2 bytes
              | (1 << 20)                            // pad_enable
              | (6 << 22)                            // pad_interval: 128 DW = 512B
              | (3 << 25);                           // pad_amount: 4 DW = 16B
        g1[1] = (int)(256u << 16);                   // tensor_dim0 = 256
        g1[2] = (int)(256u << 16);                   // tensor_dim1 = 256
        g1[3] = (int)(256u << 16);                   // tile_dim0 = 256
        g1[4] = 256;                                 // tile_dim1 = 256 (tile_dim2 = 0)
        g1[5] = 256;                                 // tensor_dim0_stride = 256
        g1[6] = 0;
        g1[7] = 0;
        v4i z4 = {0, 0, 0, 0};                       // groups 2/3 unused (2-D tile)
        v8i z8 = {0, 0, 0, 0, 0, 0, 0, 0};           // trailing group (clang-23 6-arg form)
        __builtin_amdgcn_tensor_load_to_lds(g0, g1, z4, z4, z8, 0);
        __builtin_amdgcn_s_wait_tensorcnt(0);
    }
#else
    // Fallback: cooperative b128 copy, 512B rows -> KSTR-padded LDS rows
    for (int idx = tid; idx < 256 * 32; idx += THREADS) {
        int n = idx >> 5, c = idx & 31;
        *(uint4*)(lw + n * KSTR + c * 8) = ((const uint4*)(gsrc + n * 256))[c];
    }
#endif
}

// ---------------- One 128x256 layer via v_wmma_f32_16x16x32_bf16 -------------------------------
// Wave w owns column tiles {2w, 2w+1} across all 8 row tiles (16 v8f accumulators).
__device__ __forceinline__ void wmma_layer(const uint16_t* __restrict__ lx,
                                           uint16_t* __restrict__ ly,
                                           const uint16_t* __restrict__ lw,
                                           const float* __restrict__ bias,
                                           int ksteps, int tid) {
    const int lane = tid & 31;
    const int wid  = tid >> 5;
    const int l15  = lane & 15;
    const int h    = lane >> 4;          // half-wave group
    const int n0   = (wid * 2 + 0) * 16 + l15;
    const int n1   = (wid * 2 + 1) * 16 + l15;

    v8f acc[8][2];
    const v8f vzero = {0.f, 0.f, 0.f, 0.f, 0.f, 0.f, 0.f, 0.f};
#pragma unroll
    for (int rt = 0; rt < 8; ++rt) { acc[rt][0] = vzero; acc[rt][1] = vzero; }

    for (int ks = 0; ks < ksteps; ++ks) {
        // B fragment (contiguous k order, same as A): lane = col n, k = ks*32 + h*16 + 2v + p
        // -> 16 contiguous bf16 (32B) per lane: two ds_load_b128
        Frag B0, B1;
        {
            const uint4* p0 = (const uint4*)(lw + n0 * KSTR + ks * 32 + h * 16);
            const uint4* p1 = (const uint4*)(lw + n1 * KSTR + ks * 32 + h * 16);
            *(uint4*)&B0.u[0] = p0[0];
            *(uint4*)&B0.u[4] = p0[1];
            *(uint4*)&B1.u[0] = p1[0];
            *(uint4*)&B1.u[4] = p1[1];
        }
#pragma unroll
        for (int rt = 0; rt < 8; ++rt) {
            // A fragment (matches B): row = rt*16+l15,
            // k = ks*32 + h*16 + 2v + p -> two contiguous 16B chunks
            Frag A;
            const uint4* pa = (const uint4*)(lx + (rt * 16 + l15) * XSTR + ks * 32 + h * 16);
            *(uint4*)&A.u[0] = pa[0];    // k-offsets 0..7   (+h*16)
            *(uint4*)&A.u[4] = pa[1];    // k-offsets 8..15  (+h*16)
            acc[rt][0] = __builtin_amdgcn_wmma_f32_16x16x32_bf16(
                false, A.v, false, B0.v, (short)0, acc[rt][0], false, false);
            acc[rt][1] = __builtin_amdgcn_wmma_f32_16x16x32_bf16(
                false, A.v, false, B1.v, (short)0, acc[rt][1], false, false);
        }
    }

    // Epilogue: bias + ReLU + bf16 store (C layout: lane l15 = col, row = rt*16 + j + 8h)
#pragma unroll
    for (int cc = 0; cc < 2; ++cc) {
        int   n  = (wid * 2 + cc) * 16 + l15;
        float bv = bias[n];
#pragma unroll
        for (int rt = 0; rt < 8; ++rt) {
#pragma unroll
            for (int j = 0; j < 8; ++j) {
                int   row = rt * 16 + j + 8 * h;
                float val = acc[rt][cc][j] + bv;
                val = val > 0.f ? val : 0.f;
                ly[row * XSTR + n] = f2bf(val);
            }
        }
    }
}

__global__ void __launch_bounds__(THREADS)
liif_fused_kernel(const float* __restrict__ feat, const float* __restrict__ feat_grid,
                  const float* __restrict__ s_grid, const float* __restrict__ s_cell,
                  const uint16_t* __restrict__ ws,   // pre-converted bf16 weights (4 layers)
                  const float* __restrict__ b0, const float* __restrict__ b1,
                  const float* __restrict__ b2, const float* __restrict__ b3,
                  const float* __restrict__ w4, const float* __restrict__ b4,
                  float* __restrict__ out, int N) {
    extern __shared__ char smem[];
    uint16_t* X0    = (uint16_t*)(smem + X0_OFF);
    uint16_t* X1    = (uint16_t*)(smem + X1_OFF);
    uint16_t* LW    = (uint16_t*)(smem + LW_OFF);
    float*    areas = (float*)(smem + AREA_OFF);
    float*    preds = (float*)(smem + PRED_OFF);

    const int tid = threadIdx.x;
    const int wgx = blockIdx.x;
    const int HW  = HDIM * WDIM;

    // Kick off layer-0 weight DMA early so it overlaps the gather phase.
    stage_weights_tdm(ws, LW, tid);

    // ---------------- Phase 0: gather + build bf16 input rows ----------------
    {
        int row = tid >> 1;          // 2 threads per row
        int sub = tid & 1;
        int pl  = row >> 2;          // local point
        int s   = row & 3;           // shift index: (r,c) = (s>>1, s&1)
        int n   = wgx * PTS + pl;
        if (n < N) {
            float sg0 = s_grid[n];
            float sg1 = s_grid[N + n];
            const float dh = 1.0f / HDIM, dw = 1.0f / WDIM;
            float fg0 = sg0 + (2.f * (float)(s >> 1) - 1.f) * dh;
            float fg1 = sg1 + (2.f * (float)(s & 1) - 1.f) * dw;
            fg0 = fminf(fmaxf(fg0, -1.f + 1e-6f), 1.f - 1e-6f);
            fg1 = fminf(fmaxf(fg1, -1.f + 1e-6f), 1.f - 1e-6f);
            // gx = fg1 -> W axis, gy = fg0 -> H axis; nearest, align_corners=False
            int ix = (int)rintf((fg1 + 1.f) * (float)WDIM * 0.5f - 0.5f);
            int iy = (int)rintf((fg0 + 1.f) * (float)HDIM * 0.5f - 0.5f);
            ix = min(max(ix, 0), WDIM - 1);
            iy = min(max(iy, 0), HDIM - 1);
            int base = iy * WDIM + ix;
            for (int ch = sub * 32; ch < sub * 32 + 32; ++ch)
                X0[row * XSTR + ch] = f2bf(feat[ch * HW + base]);
            if (sub == 0) {
                float qg0  = feat_grid[base];
                float qg1  = feat_grid[HW + base];
                float rel0 = (sg0 - qg0) * (float)HDIM;
                float rel1 = (sg1 - qg1) * (float)WDIM;
                float qc0  = s_cell[n] * (float)HDIM;
                float qc1  = s_cell[N + n] * (float)WDIM;
                X0[row * XSTR + 64] = f2bf(rel0);
                X0[row * XSTR + 65] = f2bf(rel1);
                X0[row * XSTR + 66] = f2bf(qc0);
                X0[row * XSTR + 67] = f2bf(qc1);
                for (int k = 68; k < 96; ++k) X0[row * XSTR + k] = 0;  // K pad
                areas[row] = fabsf(rel0 * rel1);
            }
        } else {
            for (int k = sub * 48; k < sub * 48 + 48; ++k) X0[row * XSTR + k] = 0;
            if (sub == 0) areas[row] = 0.f;
        }
    }
    __syncthreads();                                  // X0 + layer-0 weights ready

    // ---------------- Phase 1: four WMMA layers, LDS-resident ----------------
    wmma_layer(X0, X1, LW, b0, 3, tid);               // 68(pad96) -> 256
    __syncthreads();

    stage_weights_tdm(ws + 1 * WS_LAYER_ELEMS, LW, tid);
    __syncthreads();
    wmma_layer(X1, X0, LW, b1, 8, tid);               // 256 -> 256
    __syncthreads();

    stage_weights_tdm(ws + 2 * WS_LAYER_ELEMS, LW, tid);
    __syncthreads();
    wmma_layer(X0, X1, LW, b2, 8, tid);               // 256 -> 256
    __syncthreads();

    stage_weights_tdm(ws + 3 * WS_LAYER_ELEMS, LW, tid);
    __syncthreads();
    wmma_layer(X1, X0, LW, b3, 8, tid);               // 256 -> 256, result in X0
    __syncthreads();

    // ---------------- Phase 2: final 256->3 (tiny) + area blend ----------------
    float* w4l = (float*)(smem + LW_OFF);             // LW region free now
    for (int i = tid; i < 256 * 3; i += THREADS) w4l[i] = w4[i];
    __syncthreads();

    for (int idx = tid; idx < ROWS * 3; idx += THREADS) {
        int row = idx / 3, oc = idx % 3;
        float acc = b4[oc];
        for (int k = 0; k < 256; k += 8) {            // b128 LDS reads (16B aligned)
            uint4 q = *(const uint4*)(X0 + row * XSTR + k);
            const uint16_t* t = (const uint16_t*)&q;
#pragma unroll
            for (int i = 0; i < 8; ++i)
                acc += bf2f(t[i]) * w4l[(k + i) * 3 + oc];
        }
        preds[idx] = acc;
    }
    __syncthreads();

    if (tid < PTS) {
        int p = tid;
        int n = wgx * PTS + p;
        if (n < N) {
            int   base = p * 4;
            float a0 = areas[base + 0], a1 = areas[base + 1];
            float a2 = areas[base + 2], a3 = areas[base + 3];
            float tot = a0 + a1 + a2 + a3 + 1e-9f;
            float q0 = a3 / tot, q1 = a2 / tot, q2 = a1 / tot, q3 = a0 / tot;  // diagonal swap
#pragma unroll
            for (int oc = 0; oc < 3; ++oc) {
                float o = preds[(base + 0) * 3 + oc] * q0
                        + preds[(base + 1) * 3 + oc] * q1
                        + preds[(base + 2) * 3 + oc] * q2
                        + preds[(base + 3) * 3 + oc] * q3;
                out[oc * N + n] = o;   // [1, 3, N, 1]
            }
        }
    }
}

extern "C" void kernel_launch(void* const* d_in, const int* in_sizes, int n_in,
                              void* d_out, int out_size, void* d_ws, size_t ws_size,
                              hipStream_t stream) {
    const float* feat      = (const float*)d_in[0];
    const float* feat_grid = (const float*)d_in[1];
    const float* s_grid    = (const float*)d_in[2];
    const float* s_cell    = (const float*)d_in[3];
    const float* w0 = (const float*)d_in[4];
    const float* b0 = (const float*)d_in[5];
    const float* w1 = (const float*)d_in[6];
    const float* b1 = (const float*)d_in[7];
    const float* w2 = (const float*)d_in[8];
    const float* b2 = (const float*)d_in[9];
    const float* w3 = (const float*)d_in[10];
    const float* b3 = (const float*)d_in[11];
    const float* w4 = (const float*)d_in[12];
    const float* b4 = (const float*)d_in[13];
    float* out = (float*)d_out;
    uint16_t* ws = (uint16_t*)d_ws;          // 4 * 65536 bf16 = 512KB scratch

    int N = in_sizes[2] / 2;                 // s_grid is [1, 2, N, 1]
    int grid = (N + PTS - 1) / PTS;

    // Pre-pass: convert + transpose weights to bf16 once (stream-ordered).
    prep_weights_kernel<<<(4 * WS_LAYER_ELEMS) / THREADS, THREADS, 0, stream>>>(
        w0, w1, w2, w3, ws);

    // Allow > default dynamic LDS (WGP has 320KB); host-side, capture-safe.
    (void)hipFuncSetAttribute((const void*)liif_fused_kernel,
                              hipFuncAttributeMaxDynamicSharedMemorySize,
                              SMEM_BYTES);

    liif_fused_kernel<<<grid, THREADS, SMEM_BYTES, stream>>>(
        feat, feat_grid, s_grid, s_cell,
        ws, b0, b1, b2, b3, w4, b4,
        out, N);
}
